// LaCore_14070312861947
// MI455X (gfx1250) — matrix operand order repaired and checked by reading, not compiled, not run on hardware
//
#include <hip/hip_runtime.h>
#include <hip/hip_bf16.h>

#define NNODES 100000
#define NEDGES 640000
#define FDIM   128
#define HDIM   128
#define NGRAPH 64
#define NCLUST 50000
#define NCLSS  10

typedef __attribute__((ext_vector_type(16))) _Float16 v16h;
typedef __attribute__((ext_vector_type(8)))  _Float16 v8h;
typedef __attribute__((ext_vector_type(8)))  float    v8f;

// ------------------------------------------------------------------
// Edge aggregation, conv1: AGG[dst] += x[src], DEG[dst] += 1
// One 32-lane group per edge, grid-stride; prefetch next edge's row
// (global_prefetch) while current row's atomics are in flight.
// ------------------------------------------------------------------
__global__ __launch_bounds__(256) void edge_agg(
    const int* __restrict__ src, const int* __restrict__ dst,
    const float* __restrict__ X, float* __restrict__ AGG,
    float* __restrict__ DEG, int E)
{
  int lane = threadIdx.x & 31;
  int grp  = blockIdx.x * 8 + (threadIdx.x >> 5);
  int nst  = gridDim.x * 8;
  for (int e = grp; e < E; e += nst) {
    int en = e + nst;
    if (en < E) {
      int sn = src[en];
      __builtin_prefetch(X + (size_t)sn * FDIM + lane * 4, 0, 0);
    }
    int s = src[e], d = dst[e];
    float4 v = ((const float4*)(X + (size_t)s * FDIM))[lane];
    float* out = AGG + (size_t)d * FDIM + lane * 4;
    atomicAdd(out + 0, v.x);
    atomicAdd(out + 1, v.y);
    atomicAdd(out + 2, v.z);
    atomicAdd(out + 3, v.w);
    if (lane == 0) atomicAdd(&DEG[d], 1.0f);
  }
}

// Coarsened-edge aggregation (conv2/3): self-loops masked out.
__global__ __launch_bounds__(256) void edge_agg_masked(
    const int* __restrict__ src, const int* __restrict__ dst,
    const int* __restrict__ cl,
    const float* __restrict__ X, float* __restrict__ AGG,
    float* __restrict__ DEG, int E)
{
  int lane = threadIdx.x & 31;
  int grp  = blockIdx.x * 8 + (threadIdx.x >> 5);
  int nst  = gridDim.x * 8;
  for (int e = grp; e < E; e += nst) {
    int en = e + nst;
    if (en < E) {
      int pn = cl[src[en]];
      __builtin_prefetch(X + (size_t)pn * HDIM + lane * 4, 0, 0);
    }
    int ps = cl[src[e]], pd = cl[dst[e]];
    if (ps == pd) continue;  // emask
    float4 v = ((const float4*)(X + (size_t)ps * HDIM))[lane];
    float* out = AGG + (size_t)pd * HDIM + lane * 4;
    atomicAdd(out + 0, v.x);
    atomicAdd(out + 1, v.y);
    atomicAdd(out + 2, v.z);
    atomicAdd(out + 3, v.w);
    if (lane == 0) atomicAdd(&DEG[pd], 1.0f);
  }
}

// ------------------------------------------------------------------
// Fused dual GEMM + bias + ReLU (WMMA f16->f32):
//   OUT = relu( (A0 / max(deg,1)) @ Wa^T  +  A1 @ Wb^T  +  bias )
// Block = 128 threads (4 waves); each wave -> 16-row x 128-col tile.
// Weights brought in via async global->LDS (ASYNCcnt) into a dynamic
// 32KB staging buffer, then converted into a 64KB static f16 table
// with XOR-granule swizzle (conflict-free WMMA operand reads).
// ------------------------------------------------------------------
__device__ __forceinline__ v16h lds_frag(const _Float16* row, int physGran)
{
  const _Float16* p = row + physGran * 16;
  v8h lo = *(const v8h*)p;
  v8h hi = *(const v8h*)(p + 8);
  v16h r;
#pragma unroll
  for (int i = 0; i < 8; ++i) { r[i] = lo[i]; r[i + 8] = hi[i]; }
  return r;
}

__global__ __launch_bounds__(128) void dual_gemm_relu_wmma(
    const float* __restrict__ A0, const float* __restrict__ deg,
    const float* __restrict__ A1,
    const float* __restrict__ Wa, const float* __restrict__ Wb,
    const float* __restrict__ bias,
    float* __restrict__ OUT, int M)
{
  // [2][row 0..127][col 0..127], col granules of 16 halves, XOR swizzled
  __shared__ _Float16 sW[2][HDIM][HDIM];
  extern __shared__ float stage[];   // dynamic: 64 rows x 128 f32 = 32KB

  const float* Wsrc[2] = { Wa, Wb };
  uint32_t lbase = (uint32_t)(uintptr_t)stage;   // raw LDS byte address
#pragma unroll
  for (int mtx = 0; mtx < 2; ++mtx) {
#pragma unroll
    for (int chunk = 0; chunk < 2; ++chunk) {
      const float* gsrc = Wsrc[mtx] + (size_t)chunk * 64 * HDIM;
      // async copy 8192 f32 (2048 x b128), no VGPR round-trip
      for (int i = threadIdx.x; i < 2048; i += 128) {
        uint32_t laddr = lbase + i * 16;
        uint64_t gaddr = (uint64_t)(uintptr_t)(gsrc + i * 4);
        asm volatile("global_load_async_to_lds_b128 %0, %1, off"
                     :: "v"(laddr), "v"(gaddr) : "memory");
      }
      asm volatile("s_wait_asynccnt 0" ::: "memory");
      __syncthreads();
      // convert f32 -> f16 into swizzled table
      for (int t = threadIdx.x; t < 64 * HDIM; t += 128) {
        int r = chunk * 64 + (t >> 7), c = t & 127;
        int pc = ((((c >> 4) ^ (r & 7)) << 4) | (c & 15));
        sW[mtx][r][pc] = (_Float16)stage[t];
      }
      __syncthreads();
    }
  }

  int wave = threadIdx.x >> 5;
  int lane = threadIdx.x & 31;
  int hh   = lane >> 4;   // half-wave: selects K sub-range
  int lr   = lane & 15;   // row (A) / col (B,D) within 16-tile

  int m0 = blockIdx.x * 64 + wave * 16;
  int m  = m0 + lr;
  int mc = (m < M) ? m : (M - 1);           // clamp loads, mask stores
  const float* a0 = A0 + (size_t)mc * HDIM;
  const float* a1 = A1 + (size_t)mc * HDIM;
  float s0 = 1.0f / fmaxf(deg[mc], 1.0f);   // fused mean-normalization

  // Build all A fragments (4 K-steps of 32, both operand matrices).
  // A layout: elems 0..15 -> K = k0 + hh*16 + i (contiguous).
  v16h fa0[4], fa1[4];
  int aoff = hh * 16;
#pragma unroll
  for (int ks = 0; ks < 4; ++ks) {
    int k0 = ks * 32;
    v16h r0, r1;
#pragma unroll
    for (int i = 0; i < 8; ++i) {
      r0[i]     = (_Float16)(a0[k0 + aoff + i]      * s0);
      r0[i + 8] = (_Float16)(a0[k0 + aoff + 8 + i] * s0);
      r1[i]     = (_Float16)(a1[k0 + aoff + i]);
      r1[i + 8] = (_Float16)(a1[k0 + aoff + 8 + i]);
    }
    fa0[ks] = r0; fa1[ks] = r1;
  }

  // 8 output 16x16 tiles per wave; accumulate rel + root into one C.
#pragma unroll
  for (int nt = 0; nt < 8; ++nt) {
    int ncol = nt * 16 + lr;       // this lane's output column (B/D lane = N)
    v8f acc;
#pragma unroll
    for (int i = 0; i < 8; ++i) acc[i] = 0.0f;

#pragma unroll
    for (int ks = 0; ks < 4; ++ks) {
      // B layout: 16 contiguous K starting at k0 + hh*16  -> one granule
      int logGran  = ks * 2 + hh;
      int physGran = logGran ^ (ncol & 7);
      v16h wa = lds_frag(&sW[0][ncol][0], physGran);
      v16h wb = lds_frag(&sW[1][ncol][0], physGran);
      acc = __builtin_amdgcn_wmma_f32_16x16x32_f16(
              false, fa0[ks], false, wa, (short)0, acc, false, false);
      acc = __builtin_amdgcn_wmma_f32_16x16x32_f16(
              false, fa1[ks], false, wb, (short)0, acc, false, false);
    }

    float bv = bias[ncol];
    int rbase = m0 + hh * 8;       // D layout: lanes>=16 hold M rows 8..15
#pragma unroll
    for (int g = 0; g < 8; ++g) {
      int row = rbase + g;
      if (row < M)
        OUT[(size_t)row * HDIM + ncol] = fmaxf(acc[g] + bv, 0.0f);
    }
  }
}

// ------------------------------------------------------------------
// Per-group mean/max pooling (values are post-ReLU => >= 0, so integer
// atomicMax on the float bit pattern is exact with 0-init).
// ------------------------------------------------------------------
__global__ __launch_bounds__(256) void node_stats(
    const float* __restrict__ Hs, const int* __restrict__ gid,
    float* __restrict__ SUM, float* __restrict__ MX,
    float* __restrict__ CNT, int M)
{
  int i    = blockIdx.x * 8 + (threadIdx.x >> 5);
  int lane = threadIdx.x & 31;
  if (i >= M) return;
  int b = gid[i];
  float4 v = ((const float4*)(Hs + (size_t)i * HDIM))[lane];
  float* s = SUM + (size_t)b * HDIM + lane * 4;
  atomicAdd(s + 0, v.x); atomicAdd(s + 1, v.y);
  atomicAdd(s + 2, v.z); atomicAdd(s + 3, v.w);
  int* mx = (int*)(MX + (size_t)b * HDIM + lane * 4);
  atomicMax(mx + 0, __float_as_int(v.x));
  atomicMax(mx + 1, __float_as_int(v.y));
  atomicMax(mx + 2, __float_as_int(v.z));
  atomicMax(mx + 3, __float_as_int(v.w));
  if (lane == 0) atomicAdd(&CNT[b], 1.0f);
}

__global__ __launch_bounds__(256) void cluster_sum(
    const float* __restrict__ Hs, const int* __restrict__ cl,
    float* __restrict__ SUM, float* __restrict__ CNT, int M)
{
  int i    = blockIdx.x * 8 + (threadIdx.x >> 5);
  int lane = threadIdx.x & 31;
  if (i >= M) return;
  int c = cl[i];
  float4 v = ((const float4*)(Hs + (size_t)i * HDIM))[lane];
  float* s = SUM + (size_t)c * HDIM + lane * 4;
  atomicAdd(s + 0, v.x); atomicAdd(s + 1, v.y);
  atomicAdd(s + 2, v.z); atomicAdd(s + 3, v.w);
  if (lane == 0) atomicAdd(&CNT[c], 1.0f);
}

__global__ __launch_bounds__(256) void batch_max(
    const int* __restrict__ batch, const int* __restrict__ cl,
    int* __restrict__ BP, int Nn)
{
  int i = blockIdx.x * blockDim.x + threadIdx.x;
  if (i < Nn) atomicMax(&BP[cl[i]], batch[i]);
}

__global__ __launch_bounds__(256) void div_rows(
    float* __restrict__ S, const float* __restrict__ CNT, int rows)
{
  int idx = blockIdx.x * blockDim.x + threadIdx.x;
  if (idx < rows * HDIM) S[idx] *= (1.0f / fmaxf(CNT[idx >> 7], 1.0f));
}

// ------------------------------------------------------------------
// Head: z = relu([pm|px|qm|qx] @ W1^T + b1); logits + log_softmax.
// G=64 rows -> negligible FLOPs, keep it simple/scalar.
// ------------------------------------------------------------------
__global__ __launch_bounds__(128) void head1(
    const float* __restrict__ pm, const float* __restrict__ px,
    const float* __restrict__ qm, const float* __restrict__ qx,
    const float* __restrict__ W1, const float* __restrict__ b1,
    float* __restrict__ Z)
{
  int g = blockIdx.x, j = threadIdx.x;
  const float* w = W1 + (size_t)j * (4 * HDIM);
  float acc = b1[j];
#pragma unroll 4
  for (int k = 0; k < HDIM; ++k) acc += pm[g * HDIM + k] * w[k];
#pragma unroll 4
  for (int k = 0; k < HDIM; ++k) acc += px[g * HDIM + k] * w[HDIM + k];
#pragma unroll 4
  for (int k = 0; k < HDIM; ++k) acc += qm[g * HDIM + k] * w[2 * HDIM + k];
#pragma unroll 4
  for (int k = 0; k < HDIM; ++k) acc += qx[g * HDIM + k] * w[3 * HDIM + k];
  Z[g * HDIM + j] = fmaxf(acc, 0.0f);
}

__global__ __launch_bounds__(32) void head2(
    const float* __restrict__ Z, const float* __restrict__ W2,
    const float* __restrict__ b2, float* __restrict__ OUT)
{
  __shared__ float lg[NCLSS];
  int g = blockIdx.x, t = threadIdx.x;
  if (t < NCLSS) {
    const float* w = W2 + (size_t)t * HDIM;
    float a = b2[t];
    for (int k = 0; k < HDIM; ++k) a += Z[g * HDIM + k] * w[k];
    lg[t] = a;
  }
  __syncthreads();
  if (t == 0) {
    float mx = lg[0];
    for (int c = 1; c < NCLSS; ++c) mx = fmaxf(mx, lg[c]);
    float s = 0.0f;
    for (int c = 0; c < NCLSS; ++c) s += __expf(lg[c] - mx);
    float lse = mx + __logf(s);
    for (int c = 0; c < NCLSS; ++c) OUT[g * NCLSS + c] = lg[c] - lse;
  }
}

// ------------------------------------------------------------------
extern "C" void kernel_launch(void* const* d_in, const int* in_sizes, int n_in,
                              void* d_out, int out_size, void* d_ws, size_t ws_size,
                              hipStream_t stream)
{
  (void)in_sizes; (void)n_in; (void)out_size; (void)ws_size;
  const float* x      = (const float*)d_in[0];
  const int*   ei     = (const int*)d_in[1];
  const int*   src    = ei;
  const int*   dst    = ei + NEDGES;
  const int*   batch  = (const int*)d_in[2];
  const int*   clstr  = (const int*)d_in[3];
  const float* Wrel1  = (const float*)d_in[5];
  const float* brel1  = (const float*)d_in[6];
  const float* Wroot1 = (const float*)d_in[7];
  const float* Wrel2  = (const float*)d_in[8];
  const float* brel2  = (const float*)d_in[9];
  const float* Wroot2 = (const float*)d_in[10];
  const float* Wrel3  = (const float*)d_in[11];
  const float* brel3  = (const float*)d_in[12];
  const float* Wroot3 = (const float*)d_in[13];
  const float* W1     = (const float*)d_in[14];
  const float* b1     = (const float*)d_in[15];
  const float* W2     = (const float*)d_in[16];
  const float* b2     = (const float*)d_in[17];
  float* out = (float*)d_out;

  size_t off = 0;
  char* base = (char*)d_ws;
  auto carve = [&](size_t bytes) -> void* {
    void* p = base + off;
    off += (bytes + 255) & ~(size_t)255;
    return p;
  };
  float* AGG    = (float*)carve((size_t)NNODES * HDIM * 4);  // aliased: AGGP
  float* Hbuf   = (float*)carve((size_t)NNODES * HDIM * 4);  // H; HP; HP2 in 2nd half
  float* XP     = (float*)carve((size_t)NCLUST * HDIM * 4);
  float* DEG    = (float*)carve((size_t)NNODES * 4);
  float* DEGP   = (float*)carve((size_t)NCLUST * 4);
  float* CCNT   = (float*)carve((size_t)NCLUST * 4);
  int*   BATCHP = (int*)  carve((size_t)NCLUST * 4);
  float* PRESUM = (float*)carve((size_t)NGRAPH * HDIM * 4);
  float* PREMAX = (float*)carve((size_t)NGRAPH * HDIM * 4);
  float* POSTSUM= (float*)carve((size_t)NGRAPH * HDIM * 4);
  float* POSTMAX= (float*)carve((size_t)NGRAPH * HDIM * 4);
  float* GCNT   = (float*)carve((size_t)NGRAPH * 4);
  float* GPCNT  = (float*)carve((size_t)NGRAPH * 4);
  float* Z      = (float*)carve((size_t)NGRAPH * HDIM * 4);

  float* AGGP = AGG;                            // conv1 agg dead after GEMM1
  float* HP   = Hbuf;                           // H dead after phase 2/3
  float* HP2  = Hbuf + (size_t)NCLUST * HDIM;   // disjoint from HP (C = N/2)

  const size_t dynLds = 64 * HDIM * sizeof(float);   // 32KB async staging

  // ---- zero accumulators (graph-capture-safe memset nodes) ----
  hipMemsetAsync(AGG,    0, (size_t)NNODES * HDIM * 4, stream);
  hipMemsetAsync(DEG,    0, (size_t)NNODES * 4, stream);
  hipMemsetAsync(XP,     0, (size_t)NCLUST * HDIM * 4, stream);
  hipMemsetAsync(CCNT,   0, (size_t)NCLUST * 4, stream);
  hipMemsetAsync(BATCHP, 0, (size_t)NCLUST * 4, stream);
  hipMemsetAsync(PRESUM, 0, (size_t)NGRAPH * HDIM * 4, stream);
  hipMemsetAsync(PREMAX, 0, (size_t)NGRAPH * HDIM * 4, stream);
  hipMemsetAsync(POSTSUM,0, (size_t)NGRAPH * HDIM * 4, stream);
  hipMemsetAsync(POSTMAX,0, (size_t)NGRAPH * HDIM * 4, stream);
  hipMemsetAsync(GCNT,   0, (size_t)NGRAPH * 4, stream);
  hipMemsetAsync(GPCNT,  0, (size_t)NGRAPH * 4, stream);

  // ---- conv1 ----
  edge_agg<<<10000, 256, 0, stream>>>(src, dst, x, AGG, DEG, NEDGES);
  dual_gemm_relu_wmma<<<(NNODES + 63) / 64, 128, dynLds, stream>>>(
      AGG, DEG, x, Wrel1, Wroot1, brel1, Hbuf, NNODES);

  // ---- pre-pool stats + cluster pooling ----
  node_stats<<<(NNODES + 7) / 8, 256, 0, stream>>>(Hbuf, batch, PRESUM, PREMAX, GCNT, NNODES);
  cluster_sum<<<(NNODES + 7) / 8, 256, 0, stream>>>(Hbuf, clstr, XP, CCNT, NNODES);
  batch_max<<<(NNODES + 255) / 256, 256, 0, stream>>>(batch, clstr, BATCHP, NNODES);
  div_rows<<<(NGRAPH * HDIM + 255) / 256, 256, 0, stream>>>(PRESUM, GCNT, NGRAPH);
  div_rows<<<(NCLUST * HDIM + 255) / 256, 256, 0, stream>>>(XP, CCNT, NCLUST);

  // ---- conv2 (coarsened graph, self-loops masked) ----
  hipMemsetAsync(AGGP, 0, (size_t)NCLUST * HDIM * 4, stream);
  hipMemsetAsync(DEGP, 0, (size_t)NCLUST * 4, stream);
  edge_agg_masked<<<10000, 256, 0, stream>>>(src, dst, clstr, XP, AGGP, DEGP, NEDGES);
  dual_gemm_relu_wmma<<<(NCLUST + 63) / 64, 128, dynLds, stream>>>(
      AGGP, DEGP, XP, Wrel2, Wroot2, brel2, HP, NCLUST);

  // ---- conv3 ----
  hipMemsetAsync(AGGP, 0, (size_t)NCLUST * HDIM * 4, stream);
  hipMemsetAsync(DEGP, 0, (size_t)NCLUST * 4, stream);
  edge_agg_masked<<<10000, 256, 0, stream>>>(src, dst, clstr, HP, AGGP, DEGP, NEDGES);
  dual_gemm_relu_wmma<<<(NCLUST + 63) / 64, 128, dynLds, stream>>>(
      AGGP, DEGP, HP, Wrel3, Wroot3, brel3, HP2, NCLUST);

  // ---- post-pool stats ----
  node_stats<<<(NCLUST + 7) / 8, 256, 0, stream>>>(HP2, BATCHP, POSTSUM, POSTMAX, GPCNT, NCLUST);
  div_rows<<<(NGRAPH * HDIM + 255) / 256, 256, 0, stream>>>(POSTSUM, GPCNT, NGRAPH);

  // ---- classifier head ----
  head1<<<NGRAPH, 128, 0, stream>>>(PRESUM, PREMAX, POSTSUM, POSTMAX, W1, b1, Z);
  head2<<<NGRAPH, 32, 0, stream>>>(Z, W2, b2, out);
}
